// GCNEncoder_19851338842494
// MI455X (gfx1250) — compile-verified, matrix operand order checked by reading, not run on hardware
//
#include <hip/hip_runtime.h>

// ---------------------------------------------------------------------------
// GCN 2-layer encoder for MI455X (gfx1250, wave32).
//   h = x @ W   -> bf16 WMMA 16x16x32 (f32 acc), operands pre-packed to bf16
//   Ahat h + b  -> wave-per-edge gather + f32 atomic scatter (L2-resident)
// ---------------------------------------------------------------------------

typedef __attribute__((ext_vector_type(16))) __bf16 v16bf;
typedef __attribute__((ext_vector_type(8)))  __bf16 v8bf;   // 16 bytes -> b128 load
typedef __attribute__((ext_vector_type(8)))  float  v8f;

#define D_FEAT 128

// ---------------------------------------------------------------------------
// One-shot conversions
// ---------------------------------------------------------------------------
__global__ void cvt_f32_to_bf16(const float* __restrict__ in, __bf16* __restrict__ out,
                                int n) {
    int i = blockIdx.x * blockDim.x + threadIdx.x;
    if (i < n) out[i] = (__bf16)in[i];
}

// WT[c][k] = W[k][c], packed bf16 (128x128)
__global__ void transpose_w_bf16(const float* __restrict__ W, __bf16* __restrict__ WT) {
    int i = blockIdx.x * blockDim.x + threadIdx.x;   // i = c*128 + k
    int c = i >> 7;
    int k = i & (D_FEAT - 1);
    WT[i] = (__bf16)W[k * D_FEAT + c];
}

// ---------------------------------------------------------------------------
// GEMM: H[M,128] = Xb[M,128] @ WT^T  (Xb, WT packed bf16; M divisible by 16)
// block = 128 threads (4 waves); wave w computes tile (blockIdx.x, blockIdx.y*4+w)
// ---------------------------------------------------------------------------
__global__ __launch_bounds__(128)
void gemm_bf16_wmma(const __bf16* __restrict__ Xb, const __bf16* __restrict__ WT,
                    float* __restrict__ H) {
    const int wave  = threadIdx.x >> 5;
    const int lane  = threadIdx.x & 31;
    const int mTile = blockIdx.x;
    const int nTile = blockIdx.y * 4 + wave;
    const bool hi   = (lane >= 16);
    const int  l16  = lane & 15;

    const int row = mTile * 16 + l16;          // A-fragment row for this lane
    const int col = nTile * 16 + l16;          // B/C/D-fragment column for this lane

    const __bf16* xrow = Xb + (size_t)row * D_FEAT;
    const __bf16* wcol = WT + (size_t)col * D_FEAT;

    v8f acc = {};

    #pragma unroll
    for (int kk = 0; kk < D_FEAT; kk += 32) {
        const int ka0 = kk + (hi ? 8 : 0);     // first K this lane holds

        // A fragment: elems 0..7 -> K=ka0..+7 ; elems 8..15 -> K=ka0+16..+23
        v8bf alo = *reinterpret_cast<const v8bf*>(xrow + ka0);
        v8bf ahi = *reinterpret_cast<const v8bf*>(xrow + ka0 + 16);
        // B fragment: same K packing, lanes index columns (WT row = W column)
        v8bf blo = *reinterpret_cast<const v8bf*>(wcol + ka0);
        v8bf bhi = *reinterpret_cast<const v8bf*>(wcol + ka0 + 16);

        v16bf a = __builtin_shufflevector(alo, ahi, 0,1,2,3,4,5,6,7,8,9,10,11,12,13,14,15);
        v16bf b = __builtin_shufflevector(blo, bhi, 0,1,2,3,4,5,6,7,8,9,10,11,12,13,14,15);

        acc = __builtin_amdgcn_wmma_f32_16x16x32_bf16(
            /*neg_a=*/false, a, /*neg_b=*/false, b,
            /*c_mod=*/(short)0, acc, /*reuse_a=*/false, /*reuse_b=*/false);
    }

    // D: VGPR i -> row mTile*16 + i + (hi?8:0), column 'col'
    const int rbase = mTile * 16 + (hi ? 8 : 0);
    #pragma unroll
    for (int i = 0; i < 8; ++i) {
        H[(size_t)(rbase + i) * D_FEAT + col] = acc[i];
    }
}

// ---------------------------------------------------------------------------
// Degree / normalization
// ---------------------------------------------------------------------------
__global__ void deg_init(float* __restrict__ deg, int n) {
    int i = blockIdx.x * blockDim.x + threadIdx.x;
    if (i < n) deg[i] = 1.0f;                  // self-loop weight
}

__global__ void deg_edges(const int* __restrict__ dst, const float* __restrict__ ew,
                          float* __restrict__ deg, int E) {
    int e = blockIdx.x * blockDim.x + threadIdx.x;
    if (e < E) atomicAdd(&deg[dst[e]], ew[e]);
}

__global__ void deg_to_dinv(float* __restrict__ deg, int n) {
    int i = blockIdx.x * blockDim.x + threadIdx.x;
    if (i < n) {
        float d = deg[i];
        deg[i] = (d > 0.0f) ? rsqrtf(d) : 0.0f;
    }
}

// ---------------------------------------------------------------------------
// acc[n,k] = dinv[n]^2 * h[n,k] + bias[k]   (self-loop term + bias)
// ---------------------------------------------------------------------------
__global__ void agg_init(const float* __restrict__ h, const float* __restrict__ dinv,
                         const float* __restrict__ bias, float* __restrict__ acc, int n) {
    int idx = blockIdx.x * blockDim.x + threadIdx.x;
    if (idx < n * D_FEAT) {
        int node = idx >> 7;
        int k    = idx & (D_FEAT - 1);
        float di = dinv[node];
        acc[idx] = di * di * h[idx] + bias[k];
    }
}

// ---------------------------------------------------------------------------
// One wave per edge: lane handles 4 contiguous dims.
// acc[dst,:] += dinv[src]*w*dinv[dst] * h[src,:]
// ---------------------------------------------------------------------------
__global__ __launch_bounds__(256)
void agg_edges(const int* __restrict__ src, const int* __restrict__ dst,
               const float* __restrict__ ew, const float* __restrict__ dinv,
               const float* __restrict__ h, float* __restrict__ acc, int E) {
    int gtid = blockIdx.x * blockDim.x + threadIdx.x;
    int e    = gtid >> 5;                       // wave index == edge index
    int lane = threadIdx.x & 31;
    if (e >= E) return;

    int   s    = src[e];
    int   d    = dst[e];
    float norm = dinv[s] * ew[e] * dinv[d];

    const float4 hv = *reinterpret_cast<const float4*>(h + (size_t)s * D_FEAT + lane * 4);
    float* ap = acc + (size_t)d * D_FEAT + lane * 4;
    atomicAdd(ap + 0, norm * hv.x);
    atomicAdd(ap + 1, norm * hv.y);
    atomicAdd(ap + 2, norm * hv.z);
    atomicAdd(ap + 3, norm * hv.w);
}

// ---------------------------------------------------------------------------
// relu + interleaved store to out[n,k,which] (out shape [N,128,2]);
// optionally also writes bf16 activations for the next layer's GEMM.
// ---------------------------------------------------------------------------
__global__ void relu_store(const float* __restrict__ acc, __bf16* __restrict__ x_next_bf,
                           float* __restrict__ out, int n, int which) {
    int idx = blockIdx.x * blockDim.x + threadIdx.x;
    if (idx < n * D_FEAT) {
        int node = idx >> 7;
        int k    = idx & (D_FEAT - 1);
        float v  = fmaxf(acc[idx], 0.0f);
        if (x_next_bf) x_next_bf[idx] = (__bf16)v;
        out[(size_t)node * (2 * D_FEAT) + k * 2 + which] = v;
    }
}

// ---------------------------------------------------------------------------

static inline int cdiv(int a, int b) { return (a + b - 1) / b; }

extern "C" void kernel_launch(void* const* d_in, const int* in_sizes, int n_in,
                              void* d_out, int out_size, void* d_ws, size_t ws_size,
                              hipStream_t stream) {
    const float* x  = (const float*)d_in[0];
    const int*   ei = (const int*)  d_in[1];   // [2, E] flattened
    const float* ew = (const float*)d_in[2];
    const float* W1 = (const float*)d_in[3];
    const float* b1 = (const float*)d_in[4];
    const float* W2 = (const float*)d_in[5];
    const float* b2 = (const float*)d_in[6];
    float* out = (float*)d_out;

    const int N = in_sizes[0] / D_FEAT;        // 50000
    const int E = in_sizes[1] / 2;             // 800000
    const int* src = ei;
    const int* dst = ei + E;

    // workspace layout, 256B-aligned slots
    const size_t feat = (size_t)N * D_FEAT;
    char* ws = (char*)d_ws;
    size_t off = 0;
    auto alloc = [&](size_t bytes) { char* p = ws + off; off = (off + bytes + 255) & ~(size_t)255; return p; };
    float*  dinv = (float*) alloc((size_t)N * 4);
    float*  bufA = (float*) alloc(feat * 4);          // h1, later acc2
    float*  bufB = (float*) alloc(feat * 4);          // acc1, later h2
    __bf16* Xbf  = (__bf16*)alloc(feat * 2);          // bf16 x, later bf16 relu(x1)
    __bf16* WT   = (__bf16*)alloc((size_t)D_FEAT * D_FEAT * 2);  // transposed bf16 weights

    const int mTiles = N / 16;                 // 3125
    dim3 gemmGrid(mTiles, 2);                  // 2 blocks x 4 waves = 8 N-tiles

    // ---- normalization ----
    deg_init   <<<cdiv(N, 256), 256, 0, stream>>>(dinv, N);
    deg_edges  <<<cdiv(E, 256), 256, 0, stream>>>(dst, ew, dinv, E);
    deg_to_dinv<<<cdiv(N, 256), 256, 0, stream>>>(dinv, N);

    // ---- layer 1 ----
    cvt_f32_to_bf16<<<cdiv(N * D_FEAT, 256), 256, 0, stream>>>(x, Xbf, N * D_FEAT);
    transpose_w_bf16<<<cdiv(D_FEAT * D_FEAT, 256), 256, 0, stream>>>(W1, WT);
    gemm_bf16_wmma<<<gemmGrid, 128, 0, stream>>>(Xbf, WT, bufA);            // h1
    agg_init <<<cdiv(N * D_FEAT, 256), 256, 0, stream>>>(bufA, dinv, b1, bufB, N);
    agg_edges<<<cdiv(E * 32, 256), 256, 0, stream>>>(src, dst, ew, dinv, bufA, bufB, E);
    relu_store<<<cdiv(N * D_FEAT, 256), 256, 0, stream>>>(bufB, Xbf, out, N, 0); // x1 (+bf16)

    // ---- layer 2 ----
    transpose_w_bf16<<<cdiv(D_FEAT * D_FEAT, 256), 256, 0, stream>>>(W2, WT);
    gemm_bf16_wmma<<<gemmGrid, 128, 0, stream>>>(Xbf, WT, bufB);            // h2
    agg_init <<<cdiv(N * D_FEAT, 256), 256, 0, stream>>>(bufB, dinv, b2, bufA, N);
    agg_edges<<<cdiv(E * 32, 256), 256, 0, stream>>>(src, dst, ew, dinv, bufB, bufA, E);
    relu_store<<<cdiv(N * D_FEAT, 256), 256, 0, stream>>>(bufA, nullptr, out, N, 1); // x2
}
